// NeuralSynth_34256659153538
// MI455X (gfx1250) — compile-verified, matrix operand order checked by reading, not run on hardware
//
#include <hip/hip_runtime.h>
#include <hip/hip_bf16.h>
#include <math.h>

// ---------------- model constants ----------------
#define HID    512
#define NPART  64
#define FSIZE  64
#define BLKLEN 160
#define SEQL   200
#define NB     16
#define NAUDIO 32000            // SEQL * BLKLEN
#define SRATE  16000.0f
#define ROWS   (NB*SEQL)        // 3200
#define PI_F   3.14159265358979323846f
#define LOG10F 2.302585092994046f

typedef __attribute__((ext_vector_type(16))) _Float16 v16h;
typedef __attribute__((ext_vector_type(8)))  _Float16 v8h;
typedef __attribute__((ext_vector_type(8)))  float    v8f;

// f16 fragment loader, unguarded fast path: two contiguous 8-half runs
// (k = hs*8 .. +7 and k = 16+hs*8 .. +7) per CDNA5 16-bit A/B layout.
// Each run is one 16-byte global_load_b128; no conversions in the hot loop.
__device__ __forceinline__ v16h frag_main(const _Float16* __restrict__ p, int hs)
{
    v8h r0 = *(const v8h*)(p + hs * 8);
    v8h r1 = *(const v8h*)(p + 16 + hs * 8);
    v16h f;
#pragma unroll
    for (int i = 0; i < 8; ++i) { f[i] = r0[i]; f[i + 8] = r1[i]; }
    return f;
}

// K-tail loader: clamped-index loads + value select, branch-free (no EXEC div.)
__device__ __forceinline__ v16h frag_tail(const _Float16* __restrict__ row, int kk, int hs, int K)
{
    v16h f;
    _Float16 z0 = (_Float16)0.0f;
#pragma unroll
    for (int i = 0; i < 8; ++i) {
        int k0 = kk + hs * 8 + i;
        int k1 = k0 + 16;
        _Float16 v0 = row[k0 < K ? k0 : 0];
        _Float16 v1 = row[k1 < K ? k1 : 0];
        f[i]     = (k0 < K) ? v0 : z0;
        f[i + 8] = (k1 < K) ? v1 : z0;
    }
    return f;
}

// =============================================================
// WMMA GEMM:  C[M,N] = A[M,K] @ W[N,K]^T + bias[N]   (A,W in f16)
// Each wave computes a 32x64 tile: 2 A-frags x 4 B-frags ->
// 8 v_wmma_f32_16x16x32_f16 per 32-wide K step.
// Row indices are clamped (OOB rows/cols feed only unstored outputs),
// so the steady-state loop has zero conditional loads.
// =============================================================
__global__ void wmma_gemm_k(const _Float16* __restrict__ A,
                            const _Float16* __restrict__ Wt,
                            const float* __restrict__ bias,
                            float* __restrict__ C,
                            int M, int N, int K)
{
    int lane = threadIdx.x;
    int wy = blockIdx.y * blockDim.y + threadIdx.y;   // 32-row strip
    int nx = blockIdx.x;                              // 64-col strip
    if (wy * 32 >= M) return;                         // wave-uniform exit

    int r  = lane & 15;
    int hs = lane >> 4;

    int ar0 = wy * 32 + r;       if (ar0 > M - 1) ar0 = M - 1;
    int ar1 = wy * 32 + 16 + r;  if (ar1 > M - 1) ar1 = M - 1;
    const _Float16* Arow0 = A + (size_t)ar0 * K;
    const _Float16* Arow1 = A + (size_t)ar1 * K;

    const _Float16* Brow[4];
#pragma unroll
    for (int j = 0; j < 4; ++j) {
        int wr = nx * 64 + j * 16 + r;
        if (wr > N - 1) wr = N - 1;
        Brow[j] = Wt + (size_t)wr * K;
    }

    v8f acc[2][4] = {};

    int kmain = K & ~31;
    for (int kk = 0; kk < kmain; kk += 32) {
        v16h a0 = frag_main(Arow0 + kk, hs);
        v16h a1 = frag_main(Arow1 + kk, hs);
#pragma unroll
        for (int j = 0; j < 4; ++j) {
            v16h b = frag_main(Brow[j] + kk, hs);
            acc[0][j] = __builtin_amdgcn_wmma_f32_16x16x32_f16(
                            false, a0, false, b, (short)0, acc[0][j], false, false);
            acc[1][j] = __builtin_amdgcn_wmma_f32_16x16x32_f16(
                            false, a1, false, b, (short)0, acc[1][j], false, false);
        }
    }
    if (kmain < K) {
        v16h a0 = frag_tail(Arow0, kmain, hs, K);
        v16h a1 = frag_tail(Arow1, kmain, hs, K);
#pragma unroll
        for (int j = 0; j < 4; ++j) {
            v16h b = frag_tail(Brow[j], kmain, hs, K);
            acc[0][j] = __builtin_amdgcn_wmma_f32_16x16x32_f16(
                            false, a0, false, b, (short)0, acc[0][j], false, false);
            acc[1][j] = __builtin_amdgcn_wmma_f32_16x16x32_f16(
                            false, a1, false, b, (short)0, acc[1][j], false, false);
        }
    }

#pragma unroll
    for (int mi = 0; mi < 2; ++mi) {
#pragma unroll
        for (int j = 0; j < 4; ++j) {
            int n = nx * 64 + j * 16 + r;
            float bn = bias[n < N ? n : 0];
#pragma unroll
            for (int v = 0; v < 8; ++v) {
                int m = wy * 32 + mi * 16 + hs * 8 + v;
                if (m < M && n < N)
                    C[(size_t)m * N + n] = acc[mi][j][v] + bn;
            }
        }
    }
}

// f32 -> f16 conversion (weights once per call, small activations)
__global__ void cvt16_k(const float* __restrict__ src, _Float16* __restrict__ dst, int n)
{
    int i = blockIdx.x * blockDim.x + threadIdx.x;
    if (i < n) dst[i] = (_Float16)src[i];
}

// =============================================================
// Direct conv1d, K=5, pad=2: in (NB,Cin,Lin) -> out (NB,Cout,Lout)
// =============================================================
__global__ void conv1d_k(const float* __restrict__ x, const float* __restrict__ w,
                         const float* __restrict__ b, float* __restrict__ y,
                         int Cin, int Cout, int Lin, int Lout, int stride, int relu)
{
    int idx = blockIdx.x * blockDim.x + threadIdx.x;
    int total = NB * Cout * Lout;
    if (idx >= total) return;
    int t   = idx % Lout;
    int tmp = idx / Lout;
    int co  = tmp % Cout;
    int bb  = tmp / Cout;
    float acc = b[co];
    int in0 = t * stride - 2;
    for (int ci = 0; ci < Cin; ++ci) {
        const float* xr = x + ((size_t)bb * Cin + ci) * Lin;
        const float* wr = w + ((size_t)co * Cin + ci) * 5;
#pragma unroll
        for (int k = 0; k < 5; ++k) {
            int ii = in0 + k;
            if (ii >= 0 && ii < Lin) acc += xr[ii] * wr[k];
        }
    }
    if (relu) acc = fmaxf(acc, 0.0f);
    y[idx] = acc;
}

// z = exp(z_var)*eps + z_mean  from conv output (NB,32,SEQL)
__global__ void z_k(const float* __restrict__ h4, const float* __restrict__ eps,
                    float* __restrict__ z)
{
    int idx = blockIdx.x * blockDim.x + threadIdx.x;
    if (idx >= NB * SEQL * 16) return;
    int j = idx % 16;
    int s = (idx / 16) % SEQL;
    int b = idx / (16 * SEQL);
    float mean = h4[((size_t)b * 32 + j) * SEQL + s];
    float var  = h4[((size_t)b * 32 + 16 + j) * SEQL + s];
    z[idx] = __expf(var) * eps[idx] + mean;
}

// LayerNorm + ReLU, one 256-thread block per row; writes f16 for next GEMM
__global__ void ln_relu_k(const float* __restrict__ x, const float* __restrict__ g,
                          const float* __restrict__ b, _Float16* __restrict__ y,
                          int N)
{
    __shared__ float s1[256], s2[256];
    int row = blockIdx.x, tid = threadIdx.x;
    const float* xr = x + (size_t)row * N;
    float sum = 0.0f, sq = 0.0f;
    for (int i = tid; i < N; i += 256) { float v = xr[i]; sum += v; sq += v * v; }
    s1[tid] = sum; s2[tid] = sq; __syncthreads();
    for (int off = 128; off > 0; off >>= 1) {
        if (tid < off) { s1[tid] += s1[tid + off]; s2[tid] += s2[tid + off]; }
        __syncthreads();
    }
    float mean = s1[0] / N;
    float var  = s2[0] / N - mean * mean;
    float rs   = rsqrtf(var + 1e-5f);
    for (int i = tid; i < N; i += 256) {
        float v = (xr[i] - mean) * rs * g[i] + b[i];
        v = fmaxf(v, 0.0f);
        y[(size_t)row * N + i] = (_Float16)v;
    }
}

__global__ void concat3_k(const _Float16* __restrict__ a, const _Float16* __restrict__ b,
                          const _Float16* __restrict__ c, _Float16* __restrict__ o)
{
    int idx = blockIdx.x * blockDim.x + threadIdx.x;
    if (idx >= ROWS * 3 * HID) return;
    int col = idx % (3 * HID);
    int row = idx / (3 * HID);
    _Float16 v;
    if (col < HID)            v = a[(size_t)row * HID + col];
    else if (col < 2 * HID)   v = b[(size_t)row * HID + col - HID];
    else                      v = c[(size_t)row * HID + col - 2 * HID];
    o[idx] = v;
}

__global__ void fillz_k(float* pf, _Float16* ph, int n)
{
    int i = blockIdx.x * blockDim.x + threadIdx.x;
    if (i < n) { pf[i] = 0.0f; ph[i] = (_Float16)0.0f; }
}

// GRU pointwise update for step t; keeps f32 state + f16 mirror for next GEMM
__global__ void gru_update_k(const float* __restrict__ xg, const float* __restrict__ gh,
                             float* __restrict__ h, _Float16* __restrict__ h16,
                             _Float16* __restrict__ ys16, int t)
{
    int idx = blockIdx.x * blockDim.x + threadIdx.x;
    if (idx >= NB * HID) return;
    int c = idx % HID;
    int b = idx / HID;
    size_t xrow = (size_t)(b * SEQL + t) * (3 * HID);
    size_t grow = (size_t)b * (3 * HID);
    float xr = xg[xrow + c],           hr = gh[grow + c];
    float xz = xg[xrow + HID + c],     hz = gh[grow + HID + c];
    float xn = xg[xrow + 2 * HID + c], hn = gh[grow + 2 * HID + c];
    float r = 1.0f / (1.0f + __expf(-(xr + hr)));
    float z = 1.0f / (1.0f + __expf(-(xz + hz)));
    float n = tanhf(xn + r * hn);
    float hprev = h[idx];
    float hnew  = (1.0f - z) * n + z * hprev;
    h[idx]   = hnew;
    h16[idx] = (_Float16)hnew;
    ys16[(size_t)(b * SEQL + t) * HID + c] = (_Float16)hnew;
}

// mod_sigmoid in place: 2*sigmoid(x)^log(10) + 1e-7
__global__ void modsig_k(float* p, int n)
{
    int i = blockIdx.x * blockDim.x + threadIdx.x;
    if (i >= n) return;
    float s = 1.0f / (1.0f + __expf(-p[i]));
    p[i] = 2.0f * __powf(s, LOG10F) + 1e-7f;
}

__global__ void alphanorm_k(float* a)
{
    int rs = blockIdx.x * blockDim.x + threadIdx.x;
    if (rs >= ROWS) return;
    float s = 0.0f;
    for (int k = 0; k < NPART; ++k) s += a[(size_t)rs * NPART + k];
    float inv = 1.0f / s;
    for (int k = 0; k < NPART; ++k) a[(size_t)rs * NPART + k] *= inv;
}

__global__ void revmean_k(const float* __restrict__ rev, float* __restrict__ rm)
{
    int idx = threadIdx.x;
    if (idx >= NB * 2) return;
    int b = idx / 2, j = idx % 2;
    float s = 0.0f;
    for (int t = 0; t < SEQL; ++t) s += rev[(size_t)(b * SEQL + t) * 2 + j];
    rm[idx] = s / SEQL;
}

// linear upsample by 160 of a (NB,SEQL) signal; scale applied after
__global__ void up1_k(const float* __restrict__ src, float* __restrict__ dst, float scale)
{
    int idx = blockIdx.x * blockDim.x + threadIdx.x;
    if (idx >= NB * NAUDIO) return;
    int b = idx / NAUDIO, i = idx % NAUDIO;
    float pos = (i + 0.5f) / 160.0f - 0.5f;
    pos = fminf(fmaxf(pos, 0.0f), (float)(SEQL - 1));
    int i0 = (int)floorf(pos);
    int i1 = (i0 + 1 < SEQL) ? i0 + 1 : SEQL - 1;
    float w = pos - (float)i0;
    float v = src[(size_t)b * SEQL + i0] * (1.0f - w) + src[(size_t)b * SEQL + i1] * w;
    dst[idx] = v * scale;
}

// alpha upsample: (NB,SEQL,64) -> (NB,NAUDIO,64)
__global__ void upalpha_k(const float* __restrict__ al, float* __restrict__ dst)
{
    long long idx = (long long)blockIdx.x * blockDim.x + threadIdx.x;
    if (idx >= (long long)NB * NAUDIO * NPART) return;
    int k = (int)(idx % NPART);
    int i = (int)((idx / NPART) % NAUDIO);
    int b = (int)(idx / ((long long)NPART * NAUDIO));
    float pos = (i + 0.5f) / 160.0f - 0.5f;
    pos = fminf(fmaxf(pos, 0.0f), (float)(SEQL - 1));
    int i0 = (int)floorf(pos);
    int i1 = (i0 + 1 < SEQL) ? i0 + 1 : SEQL - 1;
    float w = pos - (float)i0;
    float v = al[(size_t)(b * SEQL + i0) * NPART + k] * (1.0f - w)
            + al[(size_t)(b * SEQL + i1) * NPART + k] * w;
    dst[idx] = v;
}

// phi = cumsum(2*pi*f0n) - 2*pi*f0n[0]; one block per batch, block scan
__global__ void phi_k(const float* __restrict__ f0n, float* __restrict__ phi)
{
    __shared__ float ssum[256];
    int b = blockIdx.x, tid = threadIdx.x;
    const int SEG = NAUDIO / 256;   // 125
    const float* src = f0n + (size_t)b * NAUDIO;
    float local = 0.0f;
    for (int j = 0; j < SEG; ++j) local += 2.0f * PI_F * src[tid * SEG + j];
    float mysum = local;
    ssum[tid] = local; __syncthreads();
    for (int off = 1; off < 256; off <<= 1) {
        float tv = (tid >= off) ? ssum[tid - off] : 0.0f;
        __syncthreads();
        ssum[tid] += tv;
        __syncthreads();
    }
    float run = ssum[tid] - mysum;          // exclusive prefix
    float w0  = 2.0f * PI_F * src[0];
    float* dst = phi + (size_t)b * NAUDIO;
    for (int j = 0; j < SEG; ++j) {
        run += 2.0f * PI_F * src[tid * SEG + j];
        dst[tid * SEG + j] = run - w0;
    }
}

// harmonic bank: y = amp_up * sum_k antialias*alpha_up*sin(k*phi)
__global__ void harm_k(const float* __restrict__ phi, const float* __restrict__ f0n,
                       const float* __restrict__ ampup, const float* __restrict__ alphaup,
                       float* __restrict__ y)
{
    int idx = blockIdx.x * blockDim.x + threadIdx.x;
    if (idx >= NB * NAUDIO) return;
    float p = phi[idx];
    float fn = f0n[idx];
    const float* al = alphaup + (size_t)idx * NPART;
    float s = 0.0f;
    for (int k = 1; k <= NPART; ++k) {
        float aa = ((float)k * fn < 0.5f) ? 1.0f : 0.0f;
        s += aa * al[k - 1] * __sinf((float)k * p);
    }
    y[idx] = ampup[idx] * s;
}

// windowed impulse response: hw[rs][n] = fw(n) * irfft(filter_coef)[n]
__global__ void hw_k(const float* __restrict__ fc, float* __restrict__ hw)
{
    int rs = blockIdx.x;
    int n  = threadIdx.x;    // 0..63
    const float* c = fc + (size_t)rs * 33;
    float acc = c[0];
    for (int k = 1; k < 32; ++k)
        acc += 2.0f * c[k] * __cosf(2.0f * PI_F * (float)(k * n) / 64.0f);
    acc += c[32] * __cosf(PI_F * (float)n);
    acc *= (1.0f / 64.0f);
    float fw = 0.5f * (1.0f + __cosf(2.0f * PI_F * (float)n / 64.0f));  // hann rolled by 32
    hw[(size_t)rs * 64 + n] = fw * acc;
}

// S_filt = rfft(h_w,160) * rfft(noise,160), direct DFT per (row,bin)
__global__ void sfilt_k(const float* __restrict__ noise_u, const float* __restrict__ hw,
                        float* __restrict__ outS)
{
    int idx = blockIdx.x * blockDim.x + threadIdx.x;
    if (idx >= ROWS * 81) return;
    int bin = idx % 81;
    int rs  = idx / 81;
    const float* nu = noise_u + (size_t)rs * BLKLEN;
    float nr = 0.0f, ni = 0.0f, hr = 0.0f, hi = 0.0f;
    for (int n = 0; n < BLKLEN; ++n) {
        float ang = -2.0f * PI_F * (float)bin * (float)n / 160.0f;
        float sn, cs;
        __sincosf(ang, &sn, &cs);
        float x = (nu[n] * 2.0f - 1.0f) * 0.01f;
        nr += x * cs; ni += x * sn;
        if (n < 64) {
            float hv = hw[(size_t)rs * 64 + n];
            hr += hv * cs; hi += hv * sn;
        }
    }
    outS[(size_t)idx * 2]     = hr * nr - hi * ni;
    outS[(size_t)idx * 2 + 1] = hr * ni + hi * nr;
}

__global__ void impulse_k(const float* __restrict__ rn, const float* __restrict__ rm,
                          float* __restrict__ imp)
{
    int idx = blockIdx.x * blockDim.x + threadIdx.x;
    if (idx >= NB * NAUDIO) return;
    int b = idx / NAUDIO, t = idx % NAUDIO;
    float wet   = 1.0f / (1.0f + __expf(-rm[b * 2 + 0]));
    float decay = __expf(rm[b * 2 + 1]);
    float v = (rn[t] * 2.0f - 1.0f) * __expf(-decay * (float)t / SRATE) * wet;
    if (t == 0) v = 1.0f;
    imp[idx] = v;
}

// y_rev[b,n] = sum_{m<=n} y[b,m]*imp[b,n-m] ; LDS-tiled direct conv
__global__ void revconv_k(const float* __restrict__ y, const float* __restrict__ imp,
                          float* __restrict__ yr)
{
    __shared__ float sy[256];
    __shared__ float si[512];
    int b   = blockIdx.y;
    int n0  = blockIdx.x * 256;
    int tid = threadIdx.x;
    int n   = n0 + tid;
    const float* yb = y + (size_t)b * NAUDIO;
    const float* ib = imp + (size_t)b * NAUDIO;
    float acc = 0.0f;
    for (int m0 = 0; m0 <= n0 + 255 && m0 < NAUDIO; m0 += 256) {
        __builtin_prefetch(yb + m0 + 256, 0, 0);       // global_prefetch_b8
        int mi = m0 + tid;
        sy[tid] = (mi < NAUDIO) ? yb[mi] : 0.0f;
        int d0 = n0 - m0 - 255;
        int di0 = d0 + tid;
        si[tid]       = (di0 >= 0 && di0 < NAUDIO) ? ib[di0] : 0.0f;
        int di1 = d0 + 256 + tid;
        si[256 + tid] = (di1 >= 0 && di1 < NAUDIO) ? ib[di1] : 0.0f;
        __syncthreads();
        int cnt = n - m0; if (cnt > 255) cnt = 255;
        for (int mm = 0; mm <= cnt; ++mm)
            acc += sy[mm] * si[tid - mm + 255];
        __syncthreads();
    }
    yr[(size_t)b * NAUDIO + n] = acc;
}

// =============================================================
// host-side launcher
// =============================================================
extern "C" void kernel_launch(void* const* d_in, const int* in_sizes, int n_in,
                              void* d_out, int out_size, void* d_ws, size_t ws_size,
                              hipStream_t stream)
{
    (void)in_sizes; (void)n_in; (void)out_size; (void)ws_size;

    const float* x   = (const float*)d_in[0];
    const float* f0  = (const float*)d_in[1];
    const float* lo  = (const float*)d_in[2];
    const float* eps = (const float*)d_in[3];
    const float* noise_u      = (const float*)d_in[4];
    const float* reverb_noise = (const float*)d_in[5];

    // params flattened as jax pytree (dict keys sorted) starting at index 6:
    // convs(b,w x4)=6..13, dense_alpha(b,w)=14,15, dense_amp=16,17,
    // dense_filter=18,19, dense_reverb=20,21, f0_mlp=22..33, fi_mlp=34..45,
    // gru(bhh,bih,whh,wih)=46..49, lo_mlp=50..61, z_mlp=62..73
    auto IN = [&](int i) { return (const float*)d_in[i]; };

    // ---- output regions (return order: z, y_rev, amp_up, alpha_up, S_filt) ----
    float* out        = (float*)d_out;
    float* z_out      = out;                         // 16*200*16
    float* yrev_out   = z_out + NB * SEQL * 16;      // 16*32000
    float* ampup_out  = yrev_out + NB * NAUDIO;      // 16*32000
    float* alphaup_out= ampup_out + NB * NAUDIO;     // 16*32000*64
    float* sfilt_out  = alphaup_out + (size_t)NB * NAUDIO * NPART; // 16*200*81*2

    // ---- workspace: 256-byte-granular bump allocator ----
    char* base = (char*)d_ws;
    size_t off = 0;
    auto allocb = [&](size_t bytes) { void* p = base + off; off += (bytes + 255) & ~(size_t)255; return p; };
    auto allocf = [&](size_t n) { return (float*)allocb(n * 4); };
    auto alloch = [&](size_t n) { return (_Float16*)allocb(n * 2); };

    const int T256 = 256;

    // ---- persistent region: f16 copies of all GEMM weights ----
    _Float16* W16[80] = {};
    auto cvtw = [&](int idx, size_t n) {
        W16[idx] = alloch(n);
        cvt16_k<<<(unsigned)((n + T256 - 1) / T256), T256, 0, stream>>>(IN(idx), W16[idx], (int)n);
    };
    cvtw(49, (size_t)3 * HID * 3 * HID);      // gru.wih
    cvtw(48, (size_t)3 * HID * HID);          // gru.whh
    // mlp lin.w leaves: base+1, base+5, base+9
    cvtw(23, (size_t)HID * 1);   cvtw(27, (size_t)HID * HID); cvtw(31, (size_t)HID * HID); // f0_mlp
    cvtw(51, (size_t)HID * 1);   cvtw(55, (size_t)HID * HID); cvtw(59, (size_t)HID * HID); // lo_mlp
    cvtw(63, (size_t)HID * 16);  cvtw(67, (size_t)HID * HID); cvtw(71, (size_t)HID * HID); // z_mlp
    cvtw(35, (size_t)HID * HID); cvtw(39, (size_t)HID * HID); cvtw(43, (size_t)HID * HID); // fi_mlp
    cvtw(17, (size_t)1 * HID);   cvtw(15, (size_t)NPART * HID);                            // amp, alpha
    cvtw(19, (size_t)33 * HID);  cvtw(21, (size_t)2 * HID);                                // filter, reverb

    size_t mark = off;

    // phase 1: conv stack
    float* c1 = allocf((size_t)NB * 64 * 16000);
    float* c2 = allocf((size_t)NB * 64 * 4000);
    float* c3 = allocf((size_t)NB * 64 * 1000);
    float* c4 = allocf((size_t)NB * 32 * SEQL);

    // phase 2: everything after z (reuse conv region)
    off = mark;
    _Float16* f0x16   = alloch((size_t)ROWS);
    _Float16* lox16   = alloch((size_t)ROWS);
    _Float16* z16     = alloch((size_t)ROWS * 16);
    _Float16* f0h16   = alloch((size_t)ROWS * HID);
    _Float16* loh16   = alloch((size_t)ROWS * HID);
    _Float16* zh16    = alloch((size_t)ROWS * HID);
    _Float16* xcat16  = alloch((size_t)ROWS * 3 * HID);
    _Float16* tB16    = alloch((size_t)ROWS * HID);
    _Float16* ys16    = alloch((size_t)ROWS * HID);
    _Float16* gbuf16  = alloch((size_t)ROWS * HID);
    _Float16* hst16   = alloch((size_t)NB * HID);
    float* xg   = allocf((size_t)ROWS * 3 * HID);
    float* tA   = allocf((size_t)ROWS * HID);
    float* gh   = allocf((size_t)NB * 3 * HID);
    float* hst  = allocf((size_t)NB * HID);
    float* ampb = allocf((size_t)ROWS);
    float* alb  = allocf((size_t)ROWS * NPART);
    float* fib  = allocf((size_t)ROWS * 33);
    float* reb  = allocf((size_t)ROWS * 2);
    float* rm   = allocf(64);
    float* f0up = allocf((size_t)NB * NAUDIO);
    float* phi  = allocf((size_t)NB * NAUDIO);
    float* ysig = allocf((size_t)NB * NAUDIO);
    float* imp  = allocf((size_t)NB * NAUDIO);
    float* hwb  = allocf((size_t)ROWS * 64);

    auto gemm = [&](const _Float16* Ap, const _Float16* Wp, const float* bp, float* Cp,
                    int M, int N, int K) {
        dim3 grid((N + 63) / 64, (M + 127) / 128);
        dim3 blk(32, 4);
        wmma_gemm_k<<<grid, blk, 0, stream>>>(Ap, Wp, bp, Cp, M, N, K);
    };
    auto lnr = [&](const float* xp, const float* gp, const float* bp, _Float16* yp,
                   int M, int N) {
        ln_relu_k<<<dim3(M), dim3(256), 0, stream>>>(xp, gp, bp, yp, N);
    };
    // 3-layer mlp: per layer leaves = lin.b, lin.w, ln.b, ln.g at base+4l
    auto mlp3 = [&](const _Float16* in, int Kin, int bidx, _Float16* outb) {
        const _Float16* cur = in;
        int K = Kin;
        for (int l = 0; l < 3; ++l) {
            gemm(cur, W16[bidx + 4 * l + 1], IN(bidx + 4 * l + 0), tA, ROWS, HID, K);
            _Float16* dst = (l == 2) ? outb : tB16;
            lnr(tA, IN(bidx + 4 * l + 3), IN(bidx + 4 * l + 2), dst, ROWS, HID);
            cur = dst; K = HID;
        }
    };

    // ---------- encoder convs ----------
    {
        int tot = NB * 64 * 16000;
        conv1d_k<<<(tot + T256 - 1) / T256, T256, 0, stream>>>(x, IN(7), IN(6), c1, 1, 64, NAUDIO, 16000, 2, 1);
        tot = NB * 64 * 4000;
        conv1d_k<<<(tot + T256 - 1) / T256, T256, 0, stream>>>(c1, IN(9), IN(8), c2, 64, 64, 16000, 4000, 4, 1);
        tot = NB * 64 * 1000;
        conv1d_k<<<(tot + T256 - 1) / T256, T256, 0, stream>>>(c2, IN(11), IN(10), c3, 64, 64, 4000, 1000, 4, 1);
        tot = NB * 32 * SEQL;
        conv1d_k<<<(tot + T256 - 1) / T256, T256, 0, stream>>>(c3, IN(13), IN(12), c4, 64, 32, 1000, SEQL, 5, 0);
        tot = NB * SEQL * 16;
        z_k<<<(tot + T256 - 1) / T256, T256, 0, stream>>>(c4, eps, z_out);
    }

    // ---------- activations -> f16 (after conv region is dead) ----------
    cvt16_k<<<(ROWS + T256 - 1) / T256, T256, 0, stream>>>(f0, f0x16, ROWS);
    cvt16_k<<<(ROWS + T256 - 1) / T256, T256, 0, stream>>>(lo, lox16, ROWS);
    cvt16_k<<<(ROWS * 16 + T256 - 1) / T256, T256, 0, stream>>>(z_out, z16, ROWS * 16);

    // ---------- input MLPs (WMMA) ----------
    mlp3(f0x16, 1, 22, f0h16);    // f0_mlp
    mlp3(lox16, 1, 50, loh16);    // lo_mlp
    mlp3(z16,  16, 62, zh16);     // z_mlp

    {
        int tot = ROWS * 3 * HID;
        concat3_k<<<(tot + T256 - 1) / T256, T256, 0, stream>>>(zh16, f0h16, loh16, xcat16);
    }
    // xg = xcat @ wih.T + bih  (the big 3200x1536x1536 GEMM)
    gemm(xcat16, W16[49], IN(47), xg, ROWS, 3 * HID, 3 * HID);

    // ---------- GRU (200 sequential WMMA steps) ----------
    fillz_k<<<(NB * HID + T256 - 1) / T256, T256, 0, stream>>>(hst, hst16, NB * HID);
    for (int t = 0; t < SEQL; ++t) {
        gemm(hst16, W16[48], IN(46), gh, NB, 3 * HID, HID);   // gh = h @ whh.T + bhh
        gru_update_k<<<(NB * HID + T256 - 1) / T256, T256, 0, stream>>>(xg, gh, hst, hst16, ys16, t);
    }

    // ---------- final MLP + heads ----------
    mlp3(ys16, HID, 34, gbuf16);   // fi_mlp

    gemm(gbuf16, W16[17], IN(16), ampb, ROWS, 1, HID);
    modsig_k<<<(ROWS + T256 - 1) / T256, T256, 0, stream>>>(ampb, ROWS);
    gemm(gbuf16, W16[15], IN(14), alb, ROWS, NPART, HID);
    modsig_k<<<(ROWS * NPART + T256 - 1) / T256, T256, 0, stream>>>(alb, ROWS * NPART);
    gemm(gbuf16, W16[19], IN(18), fib, ROWS, 33, HID);
    modsig_k<<<(ROWS * 33 + T256 - 1) / T256, T256, 0, stream>>>(fib, ROWS * 33);
    gemm(gbuf16, W16[21], IN(20), reb, ROWS, 2, HID);

    alphanorm_k<<<(ROWS + T256 - 1) / T256, T256, 0, stream>>>(alb);
    revmean_k<<<1, 32, 0, stream>>>(reb, rm);

    // ---------- upsampling ----------
    {
        int tot = NB * NAUDIO;
        up1_k<<<(tot + T256 - 1) / T256, T256, 0, stream>>>(f0, f0up, 1.0f / SRATE);
        up1_k<<<(tot + T256 - 1) / T256, T256, 0, stream>>>(ampb, ampup_out, 1.0f);
        long long tot2 = (long long)NB * NAUDIO * NPART;
        upalpha_k<<<(unsigned)((tot2 + T256 - 1) / T256), T256, 0, stream>>>(alb, alphaup_out);
    }

    // ---------- harmonic synthesis ----------
    phi_k<<<NB, 256, 0, stream>>>(f0up, phi);
    harm_k<<<(NB * NAUDIO + T256 - 1) / T256, T256, 0, stream>>>(phi, f0up, ampup_out, alphaup_out, ysig);

    // ---------- filtered noise spectrum ----------
    hw_k<<<ROWS, 64, 0, stream>>>(fib, hwb);
    sfilt_k<<<(ROWS * 81 + T256 - 1) / T256, T256, 0, stream>>>(noise_u, hwb, sfilt_out);

    // ---------- reverb ----------
    impulse_k<<<(NB * NAUDIO + T256 - 1) / T256, T256, 0, stream>>>(reverb_noise, rm, imp);
    {
        dim3 grid(NAUDIO / 256, NB);
        revconv_k<<<grid, 256, 0, stream>>>(ysig, imp, yrev_out);
    }
}
